// TritonQuantizedLinear_71648644432323
// MI455X (gfx1250) — matrix operand order repaired and checked by reading, not compiled, not run on hardware
//
#include <hip/hip_runtime.h>

typedef __attribute__((ext_vector_type(16))) _Float16 v16h;
typedef __attribute__((ext_vector_type(8)))  _Float16 v8h;
typedef __attribute__((ext_vector_type(8)))  float    v8f;

#define M_TOT 64
#define K_TOT 8192
#define N_TOT 8192
#define GRP   128
#define NGRP  64      // K_TOT / GRP
#define KSTEP 32      // WMMA f16 K
#define T_PER_G 4     // GRP / KSTEP

// ---------------------------------------------------------------------------
// Kernel 1: convert x (64x8192 f32, 2 MB) once to f16 (1 MB in d_ws).
// A-fragments then load ready f16 with b128, no per-k-step converts.
// ---------------------------------------------------------------------------
__global__ void qlin_cvt_kernel(const float* __restrict__ x,
                                _Float16* __restrict__ xh) {
    int i = (blockIdx.x * blockDim.x + threadIdx.x) * 8;
    float4 a = *(const float4*)(x + i);
    float4 b = *(const float4*)(x + i + 4);
    v8h h;
    h[0] = (_Float16)a.x; h[1] = (_Float16)a.y;
    h[2] = (_Float16)a.z; h[3] = (_Float16)a.w;
    h[4] = (_Float16)b.x; h[5] = (_Float16)b.y;
    h[6] = (_Float16)b.z; h[7] = (_Float16)b.w;
    *(v8h*)(xh + i) = h;
}

// ---------------------------------------------------------------------------
// Kernel 2: per-row per-group sums  xsum[m][g] = sum_{k in group g} x[m,k]
// ---------------------------------------------------------------------------
__global__ void qlin_xsum_kernel(const float* __restrict__ x,
                                 float* __restrict__ xsum) {
    int idx = blockIdx.x * blockDim.x + threadIdx.x;   // 0..4095
    int m = idx >> 6;
    int g = idx & 63;
    const float* p = x + (size_t)m * K_TOT + (size_t)g * GRP;
    float s = 0.0f;
#pragma unroll 8
    for (int j = 0; j < GRP; j += 4) {
        float4 v = *(const float4*)(p + j);
        s += v.x + v.y + v.z + v.w;
    }
    xsum[m * NGRP + g] = s;
}

// ---------------------------------------------------------------------------
// Kernel 3: seed d_out with bias + zero-point correction:
//   out[m,n] = bias[n] - sum_g scales[g,n]*zeros[g,n]*xsum[m,g]
// Rank-64 update (64x8192x64), trivial next to the main GEMM; keeps the
// hot loop free of z / xsum traffic.
// ---------------------------------------------------------------------------
__global__ void qlin_pre_kernel(const float* __restrict__ scales,
                                const float* __restrict__ zeros,
                                const float* __restrict__ bias,
                                const float* __restrict__ xsum,
                                float* __restrict__ out) {
    int idx = blockIdx.x * blockDim.x + threadIdx.x;   // 0..524287
    int m = idx >> 13;          // / 8192
    int n = idx & (N_TOT - 1);
    float acc = bias[n];
    const float* xs = xsum + m * NGRP;
#pragma unroll 4
    for (int g = 0; g < NGRP; ++g) {
        acc -= scales[(size_t)g * N_TOT + n] * zeros[(size_t)g * N_TOT + n] * xs[g];
    }
    out[idx] = acc;
}

// ---------------------------------------------------------------------------
// Kernel 4: main quantized GEMM.
// Grid: N/32 = 256 blocks of 256 threads (8 waves, 4(M) x 2(N)).
// Per group: 4x V_WMMA_F32_16X16X32_F16 on raw q (exact in f16), then
// fold the group scale:  oacc += s * gacc.  Epilogue: out += oacc.
// ---------------------------------------------------------------------------
__global__ __launch_bounds__(256, 1) void qlin_wmma_kernel(
    const _Float16* __restrict__ xh,     // [64, 8192] f16 (pre-converted)
    const int*      __restrict__ qw,     // [8192, 8192] int32 (values fit int8)
    const float*    __restrict__ scales, // [64, 8192]
    float*          __restrict__ out) {  // [64, 8192], pre-seeded by kernel 3

    const int lane = threadIdx.x & 31;
    const int wave = threadIdx.x >> 5;           // 0..7
    const int m0   = (wave & 3) * 16;            // row tile base
    const int n0   = blockIdx.x * 32 + (wave >> 2) * 16;

    const int lm = lane & 15;                    // 0..15
    const int lh = lane >> 4;                    // 0 or 1

    // A-matrix (16-bit A 16x32 layout):
    //   lanes 0-15 : M=lm, K runs [0..7]  and [16..23]
    //   lanes 16-31: M=lm, K runs [8..15] and [24..31]
    const _Float16* xrow   = xh + (size_t)(m0 + lm) * K_TOT;
    const int       ka_off = lh * 8;

    // B-matrix (16-bit B 32x16 layout):
    //   lanes 0-15 : N=lm, K = 0..7,16..23 ; lanes 16-31: N=lm, K = 8..15,24..31
    const int ncol   = n0 + lm;
    const int kb_off = lh * 8;

    v8f oacc = {};

    for (int g = 0; g < NGRP; ++g) {
        v8f gacc = {};
#pragma unroll
        for (int t = 0; t < T_PER_G; ++t) {
            const int kk = g * GRP + t * KSTEP;

            // ---- A fragment: two 16B loads of ready f16 ----
            v8h h0 = *(const v8h*)(xrow + kk + ka_off);
            v8h h1 = *(const v8h*)(xrow + kk + 16 + ka_off);
            v16h a;
#pragma unroll
            for (int j = 0; j < 8; ++j) { a[j] = h0[j]; a[8 + j] = h1[j]; }

            // ---- B fragment: 16 K-strided q values down column ncol ----
            // Non-temporal: q (256 MB) is streamed exactly once; keep the
            // 192 MB L2 for xh/scales/out which all blocks re-read.
            const int* qcol = qw + (size_t)(kk + kb_off) * N_TOT + ncol;
            v16h b;
#pragma unroll
            for (int j = 0; j < 16; ++j) {
                int qv = __builtin_nontemporal_load(qcol + (size_t)(j + (j & 8)) * N_TOT);
                b[j] = (_Float16)(short)qv;
            }

            gacc = __builtin_amdgcn_wmma_f32_16x16x32_f16(
                false, a, false, b, (short)0, gacc, false, false);
        }

        // ---- fold group scale ----
        const float s = scales[(size_t)g * N_TOT + ncol];
#pragma unroll
        for (int r = 0; r < 8; ++r) oacc[r] += s * gacc[r];
    }

    // ---- epilogue: accumulate onto pre-seeded out (bias + zero-corr) ----
    // (acc VGPR r -> row m0 + r + lh*8, per 16x16 f32 C/D layout)
#pragma unroll
    for (int r = 0; r < 8; ++r) {
        const int orow = m0 + r + lh * 8;
        float* op = out + (size_t)orow * N_TOT + ncol;
        *op = *op + oacc[r];
    }
}

// ---------------------------------------------------------------------------
extern "C" void kernel_launch(void* const* d_in, const int* in_sizes, int n_in,
                              void* d_out, int out_size, void* d_ws, size_t ws_size,
                              hipStream_t stream) {
    const float* x      = (const float*)d_in[0];
    const int*   qw     = (const int*)  d_in[1];
    const float* scales = (const float*)d_in[2];
    const float* zeros  = (const float*)d_in[3];
    const float* bias   = (const float*)d_in[4];
    // d_in[5] = group_size scalar (128), hardcoded as GRP.

    _Float16* xh   = (_Float16*)d_ws;                          // 1 MB
    float*    xsum = (float*)((char*)d_ws + (size_t)M_TOT * K_TOT * 2); // 16 KB
    float*    out  = (float*)d_out;

    // Stage 1: x -> f16 copy (A operand), 64*8192/8 lanes.
    qlin_cvt_kernel<<<(M_TOT * K_TOT / 8) / 256, 256, 0, stream>>>(x, xh);

    // Stage 2: per-group row sums of x.
    qlin_xsum_kernel<<<16, 256, 0, stream>>>(x, xsum);

    // Stage 3: seed out with bias - scale*zero*xsum correction.
    qlin_pre_kernel<<<(M_TOT * N_TOT) / 256, 256, 0, stream>>>(
        scales, zeros, bias, xsum, out);

    // Stage 4: WMMA GEMM, out += sum_g s[g,n] * (x_g . q_g).
    qlin_wmma_kernel<<<N_TOT / 32, 256, 0, stream>>>(xh, qw, scales, out);
}
